// ModifiedGNN_9938554323195
// MI455X (gfx1250) — matrix operand order repaired and checked by reading, not compiled, not run on hardware
//
#include <hip/hip_runtime.h>
#include <hip/hip_bf16.h>
#include <cstdint>

// Problem dimensions (match the reference)
constexpr int NNODES = 4096;
constexpr int NEDGES = 131072;
constexpr int NG     = 16;
constexpr int FIN    = 128;
constexpr int C1DIM  = 64;
constexpr int DM     = 128;   // embed dim
constexpr int NHEAD  = 4;     // heads per MHA == number of MHA modules
constexpr int HDIM   = 32;    // per-head dim
constexpr int HIDF   = 256;
constexpr int NCLS   = 10;

typedef __attribute__((ext_vector_type(16))) _Float16 v16h;
typedef __attribute__((ext_vector_type(8)))  _Float16 v8h;
typedef __attribute__((ext_vector_type(8)))  float    v8f;
typedef __attribute__((ext_vector_type(4)))  unsigned u32x4;
typedef __attribute__((ext_vector_type(8)))  int      i32x8;
typedef __attribute__((ext_vector_type(4)))  int      i32x4;

// ---------------------------------------------------------------------------
// Generic Y = X @ W^T (+bias)(+relu) via v_wmma_f32_16x16x32_f16.
// X: [M,K] fp32, W: [N,K] fp32 (row-major, PyTorch Linear layout).
// Output to Yh (f16) if non-null, else Yf (fp32), row stride ldy.
// K must be a multiple of 32 (true for all uses here).
//
// OOB handling: A-rows / B-cols beyond M/N are clamped to the last valid
// row/col (safe addresses, duplicated data). No fragment masking is needed:
// a duplicated A-row only affects the matching C-row and a duplicated B-col
// only the matching C-col, and those outputs are suppressed by the store
// guard. Inner loop is therefore pure b128 loads + cvt + wmma.
// Fragment layouts per CDNA5 ISA 7.12.2 (wave32).
// ---------------------------------------------------------------------------
__global__ void gemm_xwt_wmma(const float* __restrict__ X, const float* __restrict__ W,
                              const float* __restrict__ bias,
                              float* __restrict__ Yf, _Float16* __restrict__ Yh,
                              int M, int N, int K, int ldy, int relu)
{
  const int lane  = threadIdx.x & 31;
  const int wave  = threadIdx.x >> 5;
  const int mtile = blockIdx.x * 16;
  const int ntile = blockIdx.y * 64 + wave * 16;
  const int row   = lane & 15;       // A-matrix M index
  const int hi    = lane >> 4;       // half-wave select
  const int klo   = hi ? 8 : 0;      // A fragment K sub-block base
  const int kb2   = hi ? 16 : 0;     // unused here; B fragment also uses klo
  const int col   = lane & 15;       // B/C-matrix N index

  const int arow = ((mtile + row) < M) ? (mtile + row) : (M - 1);  // clamp
  const int bcol = ((ntile + col) < N) ? (ntile + col) : (N - 1);  // clamp
  const float* Xr = X + (size_t)arow * K;
  const float* Wc = W + (size_t)bcol * K;

  v8f acc = {0.f, 0.f, 0.f, 0.f, 0.f, 0.f, 0.f, 0.f};

  for (int k0 = 0; k0 < K; k0 += 32) {
    // A fragment: K runs {k0+klo .. +7} and {k0+klo+16 .. +7}  (16B-aligned)
    float fa[16], fb[16];
    *(float4*)(&fa[0])  = *(const float4*)(Xr + k0 + klo);
    *(float4*)(&fa[4])  = *(const float4*)(Xr + k0 + klo + 4);
    *(float4*)(&fa[8])  = *(const float4*)(Xr + k0 + klo + 16);
    *(float4*)(&fa[12]) = *(const float4*)(Xr + k0 + klo + 20);
    // B fragment: K runs {k0+klo .. +7} and {k0+klo+16 .. +7}  (16B-aligned)
    *(float4*)(&fb[0])  = *(const float4*)(Wc + k0 + klo);
    *(float4*)(&fb[4])  = *(const float4*)(Wc + k0 + klo + 4);
    *(float4*)(&fb[8])  = *(const float4*)(Wc + k0 + klo + 16);
    *(float4*)(&fb[12]) = *(const float4*)(Wc + k0 + klo + 20);

    v16h a, b;
#pragma unroll
    for (int i = 0; i < 16; ++i) {
      a[i] = (_Float16)fa[i];
      b[i] = (_Float16)fb[i];
    }
    acc = __builtin_amdgcn_wmma_f32_16x16x32_f16(false, a, false, b, (short)0, acc,
                                                 false, false);
  }

#pragma unroll
  for (int v = 0; v < 8; ++v) {
    int r = mtile + v + hi * 8;
    int c = ntile + col;
    if (r < M && c < N) {
      float val = acc[v] + (bias ? bias[c] : 0.f);
      if (relu) val = fmaxf(val, 0.f);
      if (Yh) Yh[(size_t)r * ldy + c] = (_Float16)val;
      else    Yf[(size_t)r * ldy + c] = val;
    }
  }
}

// ---------------------------------------------------------------------------
// Flash attention: one wave handles a 16-row query tile for one head.
// qkv: [N, 3*DM] f16 (q | k | v).  O: [N, DM] fp32.
// HD=32 == WMMA K, so S = Q.K^T is one WMMA per 16-key chunk.
// Q tile is DMA'd into LDS with the Tensor Data Mover.
// ---------------------------------------------------------------------------
__global__ void attn_flash_wmma(const _Float16* __restrict__ qkv, float* __restrict__ O)
{
  const int lane  = threadIdx.x & 31;
  const int qtile = blockIdx.x * 16;
  const int head  = blockIdx.y;
  const int row   = lane & 15;
  const int hi    = lane >> 4;
  const int klo   = hi ? 8 : 0;
  const int kb2   = hi ? 16 : 0;
  const int col   = lane & 15;
  const int qs    = 3 * DM;          // qkv row stride = 384

  __shared__ _Float16 sQ[16 * 32];
  __shared__ _Float16 sP[16 * 32];

  // ---- TDM: stage the 16x32 f16 Q tile into LDS (2D tile of the qkv tensor)
  {
    const _Float16* gq = qkv + (size_t)qtile * qs + head * HDIM;
    unsigned lds_off = (unsigned)(uintptr_t)(&sQ[0]);
    unsigned long long ga = (unsigned long long)(uintptr_t)gq;
    // D# group0: count=1 | lds_addr | global_addr[56:0] | type=2
    u32x4 g0 = { 1u, lds_off,
                 (unsigned)(ga & 0xffffffffull),
                 (unsigned)((ga >> 32) & 0x01ffffffull) | (2u << 30) };
    // D# group1: data_size=2B, tensor_dim0=384, tensor_dim1=4096,
    //            tile_dim0=32, tile_dim1=16, tensor_dim0_stride=384
    i32x8 g1 = { (int)(1u << 16),
                 (int)((unsigned)qs << 16),
                 (int)((unsigned)NNODES << 16),
                 (int)(32u << 16),
                 (int)16,
                 (int)qs,
                 0, 0 };
    i32x4 z4 = { 0, 0, 0, 0 };
#if defined(__clang_major__) && (__clang_major__ >= 23)
    i32x8 z8 = { 0, 0, 0, 0, 0, 0, 0, 0 };
    __builtin_amdgcn_tensor_load_to_lds(g0, g1, z4, z4, z8, 0);
#else
    __builtin_amdgcn_tensor_load_to_lds(g0, g1, z4, z4, 0);
#endif
    __builtin_amdgcn_s_wait_tensorcnt(0);
    asm volatile("" ::: "memory");   // LDS now holds Q tile
  }
  __syncthreads();

  v16h qa;
#pragma unroll
  for (int i = 0; i < 16; ++i) qa[i] = sQ[row * 32 + kb2 + i];

  v8f o0 = {0.f,0.f,0.f,0.f,0.f,0.f,0.f,0.f};
  v8f o1 = {0.f,0.f,0.f,0.f,0.f,0.f,0.f,0.f};
  float rmax[8], rsum[8];
#pragma unroll
  for (int v = 0; v < 8; ++v) { rmax[v] = -3.0e38f; rsum[v] = 0.f; }

  const float scale = 0.17677669529663687f;   // 1/sqrt(32)

  for (int kc = 0; kc < NNODES; kc += 32) {
    if (kc + 32 < NNODES)
      __builtin_prefetch(qkv + (size_t)(kc + 32) * qs + DM + head * HDIM, 0, 1);

    // K chunk as two B fragments: 16 contiguous halves per lane -> 2x b128
    const v8h* kp0 = (const v8h*)(qkv + (size_t)(kc + col)      * qs + DM + head * HDIM + kb2);
    const v8h* kp1 = (const v8h*)(qkv + (size_t)(kc + 16 + col) * qs + DM + head * HDIM + kb2);
    v8h k0lo = kp0[0], k0hi = kp0[1];
    v8h k1lo = kp1[0], k1hi = kp1[1];
    v16h kb0 = __builtin_shufflevector(k0lo, k0hi, 0,1,2,3,4,5,6,7,8,9,10,11,12,13,14,15);
    v16h kb1 = __builtin_shufflevector(k1lo, k1hi, 0,1,2,3,4,5,6,7,8,9,10,11,12,13,14,15);

    v8f z = {0.f,0.f,0.f,0.f,0.f,0.f,0.f,0.f};
    v8f s0 = __builtin_amdgcn_wmma_f32_16x16x32_f16(false, qa, false, kb0, (short)0, z, false, false);
    v8f s1 = __builtin_amdgcn_wmma_f32_16x16x32_f16(false, qa, false, kb1, (short)0, z, false, false);

    // online softmax per query row (rows live one-per-VGPR across 16-lane halves)
#pragma unroll
    for (int v = 0; v < 8; ++v) {
      float a0 = s0[v] * scale, a1 = s1[v] * scale;
      float m = fmaxf(a0, a1);
#pragma unroll
      for (int off = 1; off < 16; off <<= 1) m = fmaxf(m, __shfl_xor(m, off, 32));
      float nm    = fmaxf(rmax[v], m);
      float alpha = __expf(rmax[v] - nm);
      rmax[v] = nm;
      float p0 = __expf(a0 - nm), p1 = __expf(a1 - nm);
      float ps = p0 + p1;
#pragma unroll
      for (int off = 1; off < 16; off <<= 1) ps += __shfl_xor(ps, off, 32);
      rsum[v] = rsum[v] * alpha + ps;
      o0[v] *= alpha; o1[v] *= alpha;
      int r = v + hi * 8;
      sP[r * 32 + col]      = (_Float16)p0;   // C layout -> row-major LDS
      sP[r * 32 + 16 + col] = (_Float16)p1;
    }
    __syncthreads();

    // reload P as an A fragment, V chunk as two B fragments (HD split in halves)
    v16h pa, vb0, vb1;
#pragma unroll
    for (int i = 0; i < 16; ++i) pa[i] = sP[row * 32 + kb2 + i];
#pragma unroll
    for (int i = 0; i < 16; ++i) {
      vb0[i] = qkv[(size_t)(kc + kb2 + i) * qs + 2 * DM + head * HDIM + col];
      vb1[i] = qkv[(size_t)(kc + kb2 + i) * qs + 2 * DM + head * HDIM + 16 + col];
    }
    o0 = __builtin_amdgcn_wmma_f32_16x16x32_f16(false, pa, false, vb0, (short)0, o0, false, false);
    o1 = __builtin_amdgcn_wmma_f32_16x16x32_f16(false, pa, false, vb1, (short)0, o1, false, false);
    __syncthreads();
  }

#pragma unroll
  for (int v = 0; v < 8; ++v) {
    float inv = 1.f / rsum[v];
    int r = qtile + v + hi * 8;
    O[(size_t)r * DM + head * HDIM + col]      = o0[v] * inv;
    O[(size_t)r * DM + head * HDIM + 16 + col] = o1[v] * inv;
  }
}

// ---------------------------------------------------------------------------
// GCN degree / scatter / misc kernels
// ---------------------------------------------------------------------------
__global__ void k_init_deg(float* deg) {
  int i = blockIdx.x * blockDim.x + threadIdx.x;
  if (i < NNODES) deg[i] = 1.f;                 // self-loop
}
__global__ void k_accum_deg(float* deg, const int* __restrict__ dst) {
  int e = blockIdx.x * blockDim.x + threadIdx.x;
  if (e < NEDGES) atomicAdd(&deg[dst[e]], 1.f);
}
__global__ void k_dinv(float* dinv) {
  int i = blockIdx.x * blockDim.x + threadIdx.x;
  if (i < NNODES) dinv[i] = rsqrtf(dinv[i]);    // deg >= 1 always
}
__global__ void k_agg_init(float* __restrict__ out, const float* __restrict__ t,
                           const float* __restrict__ dinv, const float* __restrict__ bias, int C) {
  int n = blockIdx.x;
  float w = dinv[n] * dinv[n];                  // self-loop weight
  for (int c = threadIdx.x; c < C; c += blockDim.x)
    out[(size_t)n * C + c] = bias[c] + w * t[(size_t)n * C + c];
}
__global__ void k_agg_edges(float* __restrict__ out, const float* __restrict__ t,
                            const float* __restrict__ dinv,
                            const int* __restrict__ src, const int* __restrict__ dst, int C) {
  int e = blockIdx.x;
  int s = src[e], d = dst[e];
  float w = dinv[s] * dinv[d];
  for (int c = threadIdx.x; c < C; c += blockDim.x)
    atomicAdd(&out[(size_t)d * C + c], w * t[(size_t)s * C + c]);
}
__global__ void k_relu(float* x, int n) {
  int i = blockIdx.x * blockDim.x + threadIdx.x;
  if (i < n) x[i] = fmaxf(x[i], 0.f);
}
__global__ void k_pool_init(float* gsum, float* gcnt) {
  int i = blockIdx.x * blockDim.x + threadIdx.x;
  if (i < NG * 4 * DM) gsum[i] = 0.f;
  if (i < NG) gcnt[i] = 0.f;
}
__global__ void k_pool_accum(const float* __restrict__ hcat, const int* __restrict__ batch,
                             float* gsum, float* gcnt) {
  int n = blockIdx.x;
  int b = batch[n];
  for (int c = threadIdx.x; c < 4 * DM; c += blockDim.x)
    atomicAdd(&gsum[(size_t)b * 4 * DM + c], hcat[(size_t)n * 4 * DM + c]);
  if (threadIdx.x == 0) atomicAdd(&gcnt[b], 1.f);
}
__global__ void k_pool_div(float* gsum, const float* gcnt) {
  int i = blockIdx.x * blockDim.x + threadIdx.x;
  if (i < NG * 4 * DM) gsum[i] /= fmaxf(gcnt[i / (4 * DM)], 1.f);
}
__global__ void k_logsoftmax(const float* __restrict__ logits, float* __restrict__ out) {
  int g = blockIdx.x;
  int l = threadIdx.x;
  float v = (l < NCLS) ? logits[g * NCLS + l] : -3.0e38f;
  float m = v;
  for (int off = 1; off < 32; off <<= 1) m = fmaxf(m, __shfl_xor(m, off, 32));
  float e = (l < NCLS) ? __expf(v - m) : 0.f;
  float s = e;
  for (int off = 1; off < 32; off <<= 1) s += __shfl_xor(s, off, 32);
  if (l < NCLS) out[g * NCLS + l] = v - m - __logf(s);
}

// ---------------------------------------------------------------------------
extern "C" void kernel_launch(void* const* d_in, const int* in_sizes, int n_in,
                              void* d_out, int out_size, void* d_ws, size_t ws_size,
                              hipStream_t stream) {
  (void)in_sizes; (void)n_in; (void)out_size; (void)ws_size;
  const float* x     = (const float*)d_in[0];
  const int*   ei    = (const int*)d_in[1];
  const int*   batch = (const int*)d_in[2];
  const float* W1    = (const float*)d_in[3];
  const float* b1    = (const float*)d_in[4];
  const float* W2    = (const float*)d_in[5];
  const float* b2    = (const float*)d_in[6];
  const float* W_in  = (const float*)d_in[7];
  const float* b_in  = (const float*)d_in[8];
  const float* W_out = (const float*)d_in[9];
  const float* b_out = (const float*)d_in[10];
  const float* Wf1   = (const float*)d_in[11];
  const float* bf1   = (const float*)d_in[12];
  const float* Wf2   = (const float*)d_in[13];
  const float* bf2   = (const float*)d_in[14];
  float* out = (float*)d_out;

  const int* src = ei;
  const int* dst = ei + NEDGES;

  char* p = (char*)d_ws;
  auto adv = [&p](size_t bytes) -> char* {
    char* r = p; p += (bytes + 255) & ~(size_t)255; return r;
  };
  float*    dinv   = (float*)adv((size_t)NNODES * 4);
  float*    tbuf   = (float*)adv((size_t)NNODES * DM * 4);        // pre-agg scratch
  float*    h1     = (float*)adv((size_t)NNODES * C1DIM * 4);
  float*    h2     = (float*)adv((size_t)NNODES * DM * 4);
  _Float16* qkvh   = (_Float16*)adv((size_t)NNODES * 3 * DM * 2); // per-module, reused
  float*    obuf   = (float*)adv((size_t)NNODES * DM * 4);        // per-module, reused
  float*    hcat   = (float*)adv((size_t)NNODES * 4 * DM * 4);
  float*    gsum   = (float*)adv((size_t)NG * 4 * DM * 4);
  float*    gcnt   = (float*)adv((size_t)NG * 4);
  float*    fc1o   = (float*)adv((size_t)NG * HIDF * 4);
  float*    logits = (float*)adv((size_t)NG * 16 * 4);

  // degrees -> dinv
  k_init_deg<<<NNODES / 256, 256, 0, stream>>>(dinv);
  k_accum_deg<<<NEDGES / 256, 256, 0, stream>>>(dinv, dst);
  k_dinv<<<NNODES / 256, 256, 0, stream>>>(dinv);

  // GCN conv1: t = x @ W1^T ; aggregate ; relu
  gemm_xwt_wmma<<<dim3(NNODES / 16, 1), 128, 0, stream>>>(x, W1, nullptr, tbuf, nullptr,
                                                          NNODES, C1DIM, FIN, C1DIM, 0);
  k_agg_init<<<NNODES, 64, 0, stream>>>(h1, tbuf, dinv, b1, C1DIM);
  k_agg_edges<<<NEDGES, 64, 0, stream>>>(h1, tbuf, dinv, src, dst, C1DIM);
  k_relu<<<(NNODES * C1DIM) / 256, 256, 0, stream>>>(h1, NNODES * C1DIM);

  // GCN conv2
  gemm_xwt_wmma<<<dim3(NNODES / 16, 2), 128, 0, stream>>>(h1, W2, nullptr, tbuf, nullptr,
                                                          NNODES, DM, C1DIM, DM, 0);
  k_agg_init<<<NNODES, 128, 0, stream>>>(h2, tbuf, dinv, b2, DM);
  k_agg_edges<<<NEDGES, 128, 0, stream>>>(h2, tbuf, dinv, src, dst, DM);
  k_relu<<<(NNODES * DM) / 256, 256, 0, stream>>>(h2, NNODES * DM);

  // 4 MHA modules (sequential on stream; qkvh/obuf reused)
  for (int m = 0; m < NHEAD; ++m) {
    gemm_xwt_wmma<<<dim3(NNODES / 16, 6), 128, 0, stream>>>(
        h2, W_in + (size_t)m * 3 * DM * DM, b_in + (size_t)m * 3 * DM,
        nullptr, qkvh, NNODES, 3 * DM, DM, 3 * DM, 0);
    attn_flash_wmma<<<dim3(NNODES / 16, NHEAD), 32, 0, stream>>>(qkvh, obuf);
    gemm_xwt_wmma<<<dim3(NNODES / 16, 2), 128, 0, stream>>>(
        obuf, W_out + (size_t)m * DM * DM, b_out + (size_t)m * DM,
        hcat + (size_t)m * DM, nullptr, NNODES, DM, DM, 4 * DM, 0);
  }

  // global mean pool
  k_pool_init<<<(NG * 4 * DM + 255) / 256, 256, 0, stream>>>(gsum, gcnt);
  k_pool_accum<<<NNODES, 128, 0, stream>>>(hcat, batch, gsum, gcnt);
  k_pool_div<<<(NG * 4 * DM + 255) / 256, 256, 0, stream>>>(gsum, gcnt);

  // MLP head + log_softmax
  gemm_xwt_wmma<<<dim3(1, 4), 128, 0, stream>>>(gsum, Wf1, bf1, fc1o, nullptr,
                                                NG, HIDF, 4 * DM, HIDF, 1);
  gemm_xwt_wmma<<<dim3(1, 1), 128, 0, stream>>>(fc1o, Wf2, bf2, logits, nullptr,
                                                NG, NCLS, HIDF, NCLS, 0);
  k_logsoftmax<<<NG, 32, 0, stream>>>(logits, out);
}
